// SVDHead_24850680775351
// MI455X (gfx1250) — matrix operand order repaired and checked by reading, not compiled, not run on hardware
//
#include <hip/hip_runtime.h>
#include <hip/hip_bf16.h>

typedef _Float16 f16;
typedef __attribute__((ext_vector_type(16))) _Float16 v16h;
typedef __attribute__((ext_vector_type(8)))  _Float16 v8h;
typedef __attribute__((ext_vector_type(8)))  float    v8f;

#define WMMA_F16(a,b,c) __builtin_amdgcn_wmma_f32_16x16x32_f16(false,(a),false,(b),(short)0,(c),false,false)

// ---------------- Kernel 1: cast f_tgt f32 -> f16 ----------------
__global__ __launch_bounds__(256) void cast_f16_kernel(const float* __restrict__ src,
                                                       f16* __restrict__ dst, int n) {
  int i = blockIdx.x * blockDim.x + threadIdx.x;
  int stride = gridDim.x * blockDim.x;
  for (; i < n; i += stride) dst[i] = (f16)src[i];
}

// ---------------- Kernel 2: f_src_att = f_src @ W^T + b  (f16 out) ----------------
// out[n][t] = sum_d f_src[n][d] * W[t][d] + b[t];  16384x256, K=256
__global__ __launch_bounds__(256) void linear_kernel(const float* __restrict__ fsrc,
                                                     const float* __restrict__ W,
                                                     const float* __restrict__ bias,
                                                     f16* __restrict__ out) {
  const int wave = threadIdx.x >> 5;
  const int lane = threadIdx.x & 31;
  const int hi   = lane >> 4;
  const int lm   = lane & 15;
  const int tile = blockIdx.x * 8 + wave;   // 16384 tiles
  const int rt    = tile >> 4;              // row tile (0..1023)
  const int ctile = tile & 15;              // col tile (0..15)

  const size_t arow = (size_t)rt * 16 + lm;     // f_src row (B*N flattened)
  const size_t brow = (size_t)ctile * 16 + lm;  // W row (t)
  v8f c = {};
  #pragma unroll
  for (int ks = 0; ks < 8; ++ks) {
    const int kb = ks * 32;
    // A half layout: h<8 -> K = kb + 8*hi + h ; h>=8 -> K = kb + 16 + 8*hi + (h-8)
    v8f a0 = *(const v8f*)(fsrc + arow * 256 + kb + 8 * hi);
    v8f a1 = *(const v8f*)(fsrc + arow * 256 + kb + 16 + 8 * hi);
    v16h a;
    #pragma unroll
    for (int i = 0; i < 8; ++i) { a[i] = (f16)a0[i]; a[8 + i] = (f16)a1[i]; }
    // B half layout: h<8 -> K = kb + 8*hi + h ; h>=8 -> K = kb + 16 + 8*hi + (h-8), N = lm
    v8f b0 = *(const v8f*)(W + brow * 256 + kb + 8 * hi);
    v8f b1 = *(const v8f*)(W + brow * 256 + kb + 16 + 8 * hi);
    v16h bm;
    #pragma unroll
    for (int i = 0; i < 8; ++i) { bm[i] = (f16)b0[i]; bm[8 + i] = (f16)b1[i]; }
    c = WMMA_F16(a, bm, c);
  }
  const float bb = bias[brow];
  #pragma unroll
  for (int v = 0; v < 8; ++v) {
    size_t orow = (size_t)rt * 16 + v + 8 * hi;   // C layout: M = v + 8*hi, N = lm
    out[orow * 256 + ctile * 16 + lm] = (f16)(c[v] + bb);
  }
}

// ---------------- Kernel 3: fused affinity -> online softmax -> p_corr ----------------
// Per-lane flash softmax: each lane owns one column per 16-col tile (256 cols total),
// keeps private (max,sum,acc) state; the 16 lane-states per row merge once at the end.
__global__ __launch_bounds__(256) void attn_kernel(const f16* __restrict__ fatt,
                                                   const f16* __restrict__ ftgt,
                                                   const float* __restrict__ ptgt,
                                                   float* __restrict__ pcorr) {
  const int wave = threadIdx.x >> 5;
  const int lane = threadIdx.x & 31;
  const int hi   = lane >> 4;
  const int lm   = lane & 15;
  const int rt   = blockIdx.x * 8 + wave;   // row tile 0..1023 (8 tiles/block stay in-batch)
  const int b    = (rt * 16) >> 12;         // batch = row/4096
  const size_t arow = (size_t)rt * 16 + lm;

  // Preload the whole 16x256 A panel (f16) for this row tile: 8 k-steps, 64 VGPRs
  v16h A[8];
  #pragma unroll
  for (int ks = 0; ks < 8; ++ks) {
    v8h a0 = *(const v8h*)(fatt + arow * 256 + ks * 32 + 8 * hi);
    v8h a1 = *(const v8h*)(fatt + arow * 256 + ks * 32 + 16 + 8 * hi);
    #pragma unroll
    for (int i = 0; i < 8; ++i) { A[ks][i] = a0[i]; A[ks][8 + i] = a1[i]; }
  }

  float rmax[8], rsum[8], ax[8], ay[8], az[8];
  #pragma unroll
  for (int v = 0; v < 8; ++v) { rmax[v] = -1e30f; rsum[v] = 0.f; ax[v] = 0.f; ay[v] = 0.f; az[v] = 0.f; }

  const size_t btbase = (size_t)b * 4096;
  for (int ct = 0; ct < 256; ++ct) {
    const int m0 = ct * 16;
    const size_t brow = (btbase + m0 + lm) * 256;
    if (ct + 1 < 256) {  // prefetch next tile's B row (global_prefetch_b8)
      __builtin_prefetch(ftgt + brow + 16 * 256, 0, 0);
    }
    v8f c = {};
    #pragma unroll
    for (int ks = 0; ks < 8; ++ks) {
      v8h b0 = *(const v8h*)(ftgt + brow + ks * 32 + 8 * hi);
      v8h b1 = *(const v8h*)(ftgt + brow + ks * 32 + 16 + 8 * hi);
      v16h bm;
      #pragma unroll
      for (int i = 0; i < 8; ++i) { bm[i] = b0[i]; bm[8 + i] = b1[i]; }
      c = WMMA_F16(A[ks], bm, c);
    }
    // this lane's column m = m0 + lm
    const float* pt = ptgt + (btbase + m0 + lm) * 3;
    const float px = pt[0], py = pt[1], pz = pt[2];
    // Per-lane online softmax update (no cross-lane traffic).
    #pragma unroll
    for (int v = 0; v < 8; ++v) {
      float t  = c[v];
      float nm = fmaxf(rmax[v], t);
      float sc = __expf(rmax[v] - nm);   // TRANS pipe, co-executes with XDL WMMA
      float p  = __expf(t - nm);
      rsum[v] = rsum[v] * sc + p;
      ax[v]   = ax[v] * sc + p * px;
      ay[v]   = ay[v] * sc + p * py;
      az[v]   = az[v] * sc + p * pz;
      rmax[v] = nm;
    }
  }
  // Merge the 16 per-lane states per row, then store p_corr.
  #pragma unroll
  for (int v = 0; v < 8; ++v) {
    float M = rmax[v];
    #pragma unroll
    for (int m = 8; m >= 1; m >>= 1) M = fmaxf(M, __shfl_xor(M, m, 32)); // 16-lane group max
    float f = __expf(rmax[v] - M);
    float s = rsum[v] * f, x = ax[v] * f, y = ay[v] * f, z = az[v] * f;
    #pragma unroll
    for (int m = 8; m >= 1; m >>= 1) {
      s += __shfl_xor(s, m, 32);
      x += __shfl_xor(x, m, 32);
      y += __shfl_xor(y, m, 32);
      z += __shfl_xor(z, m, 32);
    }
    if (lm == 0) {
      size_t row = (size_t)rt * 16 + v + 8 * hi;
      float inv = 1.0f / s;
      pcorr[row * 3 + 0] = x * inv;
      pcorr[row * 3 + 1] = y * inv;
      pcorr[row * 3 + 2] = z * inv;
    }
  }
}

// ---------------- Kernel 4: centroids + H + 3x3 SVD + transform ----------------
__global__ __launch_bounds__(256) void svd_kernel(const float* __restrict__ psrc,
                                                  const float* __restrict__ pcorr,
                                                  float* __restrict__ out) {
  const int b = blockIdx.x;
  const int tid = threadIdx.x;
  __shared__ float red[256];
  __shared__ float stats[15]; // cs[3], cc[3], H[9]

  const float* ps = psrc + (size_t)b * 4096 * 3;
  const float* pc = pcorr + (size_t)b * 4096 * 3;

  float ls[3] = {0.f, 0.f, 0.f}, lc[3] = {0.f, 0.f, 0.f};
  for (int n = tid; n < 4096; n += 256) {
    for (int k = 0; k < 3; ++k) { ls[k] += ps[n * 3 + k]; lc[k] += pc[n * 3 + k]; }
  }
  for (int q = 0; q < 6; ++q) {
    red[tid] = (q < 3) ? ls[q] : lc[q - 3];
    __syncthreads();
    for (int off = 128; off > 0; off >>= 1) {
      if (tid < off) red[tid] += red[tid + off];
      __syncthreads();
    }
    if (tid == 0) stats[q] = red[0] * (1.0f / 4096.0f);
    __syncthreads();
  }
  const float cs0 = stats[0], cs1 = stats[1], cs2 = stats[2];
  const float cc0 = stats[3], cc1 = stats[4], cc2 = stats[5];

  float h[9] = {0.f, 0.f, 0.f, 0.f, 0.f, 0.f, 0.f, 0.f, 0.f};
  for (int n = tid; n < 4096; n += 256) {
    float a0 = ps[n * 3 + 0] - cs0, a1 = ps[n * 3 + 1] - cs1, a2 = ps[n * 3 + 2] - cs2;
    float b0 = pc[n * 3 + 0] - cc0, b1 = pc[n * 3 + 1] - cc1, b2 = pc[n * 3 + 2] - cc2;
    h[0] += a0 * b0; h[1] += a0 * b1; h[2] += a0 * b2;
    h[3] += a1 * b0; h[4] += a1 * b1; h[5] += a1 * b2;
    h[6] += a2 * b0; h[7] += a2 * b1; h[8] += a2 * b2;
  }
  for (int q = 0; q < 9; ++q) {
    red[tid] = h[q];
    __syncthreads();
    for (int off = 128; off > 0; off >>= 1) {
      if (tid < off) red[tid] += red[tid + off];
      __syncthreads();
    }
    if (tid == 0) stats[6 + q] = red[0];
    __syncthreads();
  }

  if (tid == 0) {
    float H[3][3] = {{stats[6], stats[7], stats[8]},
                     {stats[9], stats[10], stats[11]},
                     {stats[12], stats[13], stats[14]}};
    // S = H^T H (symmetric), Jacobi eigen-decomposition -> V, lambda
    float S[3][3];
    for (int i = 0; i < 3; ++i)
      for (int j = 0; j < 3; ++j)
        S[i][j] = H[0][i] * H[0][j] + H[1][i] * H[1][j] + H[2][i] * H[2][j];
    float Vm[3][3] = {{1.f, 0.f, 0.f}, {0.f, 1.f, 0.f}, {0.f, 0.f, 1.f}};
    const int PP[3] = {0, 0, 1}, QQ[3] = {1, 2, 2};
    for (int sweep = 0; sweep < 12; ++sweep) {
      for (int r = 0; r < 3; ++r) {
        int p = PP[r], q = QQ[r];
        float apq = S[p][q];
        if (fabsf(apq) > 1e-20f) {
          float th = 0.5f * atan2f(2.f * apq, S[q][q] - S[p][p]);
          float c = cosf(th), s = sinf(th);
          for (int k = 0; k < 3; ++k) {  // S <- S*G
            float t0 = c * S[k][p] - s * S[k][q];
            float t1 = s * S[k][p] + c * S[k][q];
            S[k][p] = t0; S[k][q] = t1;
          }
          for (int k = 0; k < 3; ++k) {  // S <- G^T*S
            float t0 = c * S[p][k] - s * S[q][k];
            float t1 = s * S[p][k] + c * S[q][k];
            S[p][k] = t0; S[q][k] = t1;
          }
          for (int k = 0; k < 3; ++k) {  // V <- V*G
            float t0 = c * Vm[k][p] - s * Vm[k][q];
            float t1 = s * Vm[k][p] + c * Vm[k][q];
            Vm[k][p] = t0; Vm[k][q] = t1;
          }
        }
      }
    }
    float lam[3] = {S[0][0], S[1][1], S[2][2]};
    int idx[3] = {0, 1, 2};
    if (lam[idx[0]] < lam[idx[1]]) { int t = idx[0]; idx[0] = idx[1]; idx[1] = t; }
    if (lam[idx[1]] < lam[idx[2]]) { int t = idx[1]; idx[1] = idx[2]; idx[2] = t; }
    if (lam[idx[0]] < lam[idx[1]]) { int t = idx[0]; idx[0] = idx[1]; idx[1] = t; }

    float Vs[3][3], Us[3][3];
    for (int cN = 0; cN < 3; ++cN) {
      int j = idx[cN];
      float sig = sqrtf(fmaxf(lam[j], 0.f));
      for (int i = 0; i < 3; ++i) Vs[i][cN] = Vm[i][j];
      float inv = 1.0f / (sig + 1e-30f);
      for (int i = 0; i < 3; ++i)
        Us[i][cN] = (H[i][0] * Vs[0][cN] + H[i][1] * Vs[1][cN] + H[i][2] * Vs[2][cN]) * inv;
    }
    float R[3][3];
    for (int i = 0; i < 3; ++i)
      for (int j = 0; j < 3; ++j)
        R[i][j] = Vs[i][0] * Us[j][0] + Vs[i][1] * Us[j][1] + Vs[i][2] * Us[j][2];
    float det = R[0][0] * (R[1][1] * R[2][2] - R[1][2] * R[2][1]) -
                R[0][1] * (R[1][0] * R[2][2] - R[1][2] * R[2][0]) +
                R[0][2] * (R[1][0] * R[2][1] - R[1][1] * R[2][0]);
    if (det < 0.f) {
      for (int i = 0; i < 3; ++i) Vs[i][2] = -Vs[i][2];  // flip smallest-sigma right vector
      for (int i = 0; i < 3; ++i)
        for (int j = 0; j < 3; ++j)
          R[i][j] = Vs[i][0] * Us[j][0] + Vs[i][1] * Us[j][1] + Vs[i][2] * Us[j][2];
    }
    float cs[3] = {cs0, cs1, cs2}, cc[3] = {cc0, cc1, cc2};
    float* o = out + (size_t)b * 16;
    for (int i = 0; i < 3; ++i) {
      float ti = cc[i] - (R[i][0] * cs[0] + R[i][1] * cs[1] + R[i][2] * cs[2]);
      o[i * 4 + 0] = R[i][0]; o[i * 4 + 1] = R[i][1]; o[i * 4 + 2] = R[i][2]; o[i * 4 + 3] = ti;
    }
    o[12] = 0.f; o[13] = 0.f; o[14] = 0.f; o[15] = 1.f;
  }
}

extern "C" void kernel_launch(void* const* d_in, const int* in_sizes, int n_in,
                              void* d_out, int out_size, void* d_ws, size_t ws_size,
                              hipStream_t stream) {
  const float* p_src = (const float*)d_in[0];
  const float* f_src = (const float*)d_in[1];
  const float* p_tgt = (const float*)d_in[2];
  const float* f_tgt = (const float*)d_in[3];
  const float* W     = (const float*)d_in[4];
  const float* bias  = (const float*)d_in[5];
  float* out = (float*)d_out;

  // Workspace layout: fatt16 (8.39 MB) | ftgt16 (8.39 MB) | p_corr (196 KB)
  char* ws = (char*)d_ws;
  f16*   fatt16 = (f16*)ws;
  f16*   ftgt16 = (f16*)(ws + 16384ull * 256 * 2);
  float* pcorr  = (float*)(ws + 2ull * 16384 * 256 * 2);

  cast_f16_kernel<<<1024, 256, 0, stream>>>(f_tgt, ftgt16, 16384 * 256);
  linear_kernel<<<2048, 256, 0, stream>>>(f_src, W, bias, fatt16);   // 16384 tiles, 8 waves/block
  attn_kernel<<<128, 256, 0, stream>>>(fatt16, ftgt16, p_tgt, pcorr); // 1024 row tiles
  svd_kernel<<<4, 256, 0, stream>>>(p_src, pcorr, out);
}
